// DotProductAttention_9749575762784
// MI455X (gfx1250) — matrix operand order repaired and checked by reading, not compiled, not run on hardware
//
#include <hip/hip_runtime.h>

#define B_    8
#define N_    4096
#define M_    4096
#define D_    64
#define STR   80          // padded LDS row stride in f16 (160 B, multiple of 16 B)
#define QROWS 128         // query rows per block
#define KTILE 64          // keys per iteration
#define NWAVES 8
#define THREADS 256
#define NTILES (M_ / KTILE)

typedef _Float16 h8   __attribute__((ext_vector_type(8)));
typedef _Float16 v16h __attribute__((ext_vector_type(16)));
typedef float    v8f  __attribute__((ext_vector_type(8)));
typedef int      v4i_gcc __attribute__((vector_size(16)));  // matches builtin param type

#if defined(__has_builtin)
#if __has_builtin(__builtin_amdgcn_global_load_async_to_lds_b128) && \
    __has_builtin(__builtin_amdgcn_s_wait_asynccnt)
#define HAVE_ASYNC 1
#endif
#endif

union FragU { v16h v; h8 h[2]; };

// A-matrix 16x32 f16 layout: lane half hi holds K = 8*hi+(0..7) in v0-3 and 16+8*hi+(0..7) in v4-7
__device__ __forceinline__ v16h load_fragA(const _Float16* rowbase, int hi) {
  FragU f;
  f.h[0] = *(const h8*)(rowbase + 8 * hi);
  f.h[1] = *(const h8*)(rowbase + 16 + 8 * hi);
  return f.v;
}
// B-matrix 32x16 f16 layout: lane half hi holds K = 8*hi+(0..7) in v0-3 and 16+8*hi+(0..7) in v4-7
__device__ __forceinline__ v16h load_fragB(const _Float16* rowbase, int hi) {
  const _Float16* p = rowbase + 8 * hi;
  FragU f;
  f.h[0] = *(const h8*)(p);
  f.h[1] = *(const h8*)(p + 16);
  return f.v;
}

__device__ __forceinline__ float red_max16(float v) {
  v = fmaxf(v, __shfl_xor(v, 1));
  v = fmaxf(v, __shfl_xor(v, 2));
  v = fmaxf(v, __shfl_xor(v, 4));
  v = fmaxf(v, __shfl_xor(v, 8));
  return v;
}

// 16-byte global -> LDS copy; async DMA (ASYNCcnt) when available
__device__ __forceinline__ void cp16_g2l(const _Float16* g, _Float16* l) {
#ifdef HAVE_ASYNC
  __builtin_amdgcn_global_load_async_to_lds_b128(
      (__attribute__((address_space(1))) v4i_gcc*)g,
      (__attribute__((address_space(3))) v4i_gcc*)l, 0, 0);
#else
  *(uint4*)l = *(const uint4*)g;
#endif
}
__device__ __forceinline__ void wait_async0() {
#ifdef HAVE_ASYNC
  __builtin_amdgcn_s_wait_asynccnt(0);
#endif
}

// ---------------- prep: K -> f16 dense, V -> f16 transposed per 64-key tile ----
__global__ __launch_bounds__(256)
void attn_prep_kv(const float* __restrict__ Kg, const float* __restrict__ Vg,
                  _Float16* __restrict__ Kh, _Float16* __restrict__ Vt) {
  size_t idx = (size_t)blockIdx.x * 256 + threadIdx.x;  // vec4 id
  if (idx >= (size_t)B_ * M_ * D_ / 4) return;
  if (blockIdx.y == 0) {
    float4 v = *(const float4*)(Kg + idx * 4);
    _Float16* d = Kh + idx * 4;
    d[0] = (_Float16)v.x; d[1] = (_Float16)v.y;
    d[2] = (_Float16)v.z; d[3] = (_Float16)v.w;
  } else {
    // Vt layout: [b][tile][d][mloc], 4 consecutive mloc per thread (coalesced store)
    size_t o = idx * 4;
    int mloc = (int)(o & 63);
    int dd   = (int)((o >> 6) & 63);
    int tile = (int)((o >> 12) & 63);
    int b    = (int)(o >> 18);
    const float* src = Vg + ((size_t)b * M_ + (size_t)tile * 64 + mloc) * D_ + dd;
    _Float16* dst = Vt + o;
    dst[0] = (_Float16)src[0];
    dst[1] = (_Float16)src[64];
    dst[2] = (_Float16)src[128];
    dst[3] = (_Float16)src[192];
  }
}

// ---------------- main: flash attention, async double-buffered f16 tiles ------
// LDS map (f16 units): K/V double buffers + per-wave P scratch; Q staging region
// overlaps the buffers (Q is consumed into registers before tile 0 is loaded).
#define KB0 0
#define VB0 5120
#define KB1 10240
#define VB1 15360
#define PB  20480
#define SMEM_H 30720   // 61440 bytes

__global__ __launch_bounds__(THREADS)
void attn_main(const float* __restrict__ Qg, const _Float16* __restrict__ Khg,
               const _Float16* __restrict__ Vtg, float* __restrict__ Og) {
  __shared__ _Float16 sMem[SMEM_H];

  const int tid  = threadIdx.x;
  const int lane = tid & 31;
  const int wave = tid >> 5;
  const int hi   = lane >> 4;
  const int r    = lane & 15;

  const int b     = blockIdx.y;
  const int qBase = blockIdx.x * QROWS;

  const float* Qp = Qg + ((size_t)b * N_ + qBase) * D_;
  const _Float16* Khp = Khg + (size_t)b * M_ * D_;   // [m][d] dense
  const _Float16* Vtp = Vtg + (size_t)b * M_ * D_;   // [tile][d][mloc] dense

  const float QSCALE = 0.125f * 1.44269504088896340736f;  // 1/sqrt(64) * log2(e)
  const v8f zero8 = {0.f, 0.f, 0.f, 0.f, 0.f, 0.f, 0.f, 0.f};

  v16h ones;
#pragma unroll
  for (int e = 0; e < 16; ++e) ones[e] = (_Float16)1.0f;

  // ---- stage Q once into (reused) LDS, consume into register fragments ----
  for (int idx = tid; idx < QROWS * (D_ / 4); idx += THREADS) {
    int row = idx >> 4, c4 = (idx & 15) << 2;
    float4 v = *(const float4*)(Qp + (size_t)row * D_ + c4);
    _Float16* d = &sMem[row * STR + c4];
    d[0] = (_Float16)(v.x * QSCALE);
    d[1] = (_Float16)(v.y * QSCALE);
    d[2] = (_Float16)(v.z * QSCALE);
    d[3] = (_Float16)(v.w * QSCALE);
  }
  __syncthreads();
  const _Float16* qrow = &sMem[(wave * 16 + r) * STR];
  v16h qf0 = load_fragA(qrow, hi);
  v16h qf1 = load_fragA(qrow + 32, hi);
  __syncthreads();  // Q region now dead; safe to overwrite with K/V buffers

  float m_run[8], l_run[8];
  v8f o_acc[4];
#pragma unroll
  for (int j = 0; j < 8; ++j) { m_run[j] = -3.0e38f; l_run[j] = 0.f; }
#pragma unroll
  for (int g = 0; g < 4; ++g) o_acc[g] = zero8;

  // issue tile 'it' into buffer 'sel' (4 x 16B async copies per thread)
  auto issue_tile = [&](int it, int sel) {
    const _Float16* ks = Khp + (size_t)it * KTILE * D_;
    const _Float16* vs = Vtp + (size_t)it * KTILE * D_;
    _Float16* kd = &sMem[sel ? KB1 : KB0];
    _Float16* vd = &sMem[sel ? VB1 : VB0];
#pragma unroll
    for (int i = 0; i < 2; ++i) {
      int c = tid + THREADS * i;          // 0..511 chunk id
      int row = c >> 3, col = (c & 7) * 8;
      cp16_g2l(ks + row * D_ + col, kd + row * STR + col);
      cp16_g2l(vs + row * D_ + col, vd + row * STR + col);
    }
  };

  issue_tile(0, 0);

#pragma unroll 1
  for (int it = 0; it < NTILES; ++it) {
    wait_async0();       // my portion of tile 'it' landed in LDS
    __syncthreads();     // whole tile landed; previous compute done everywhere
    if (it + 1 < NTILES) issue_tile(it + 1, (it + 1) & 1);

    const _Float16* sK  = &sMem[(it & 1) ? KB1 : KB0];
    const _Float16* sVt = &sMem[(it & 1) ? VB1 : VB0];
    _Float16* myP = &sMem[PB + wave * 16 * STR];

    // ---- S = Q * K^T ----
    v8f s[4];
#pragma unroll
    for (int t = 0; t < 4; ++t) {
      const _Float16* krow = &sK[(16 * t + r) * STR];
      v16h kf0 = load_fragB(krow, hi);
      v16h kf1 = load_fragB(krow + 32, hi);
      v8f c = zero8;
      c = __builtin_amdgcn_wmma_f32_16x16x32_f16(false, qf0, false, kf0, (short)0, c, false, false);
      c = __builtin_amdgcn_wmma_f32_16x16x32_f16(false, qf1, false, kf1, (short)0, c, false, false);
      s[t] = c;
    }

    // ---- online softmax: row max via lane shuffles (exp2 domain) ----
    float mnew[8], sc[8];
#pragma unroll
    for (int j = 0; j < 8; ++j) {
      float v = fmaxf(fmaxf(s[0][j], s[1][j]), fmaxf(s[2][j], s[3][j]));
      v = red_max16(v);
      mnew[j] = fmaxf(m_run[j], v);
      sc[j]   = exp2f(m_run[j] - mnew[j]);
      m_run[j] = mnew[j];
    }
#pragma unroll
    for (int t = 0; t < 4; ++t) {
#pragma unroll
      for (int j = 0; j < 8; ++j) {
        float p = exp2f(s[t][j] - mnew[j]);
        myP[(j + 8 * hi) * STR + 16 * t + r] = (_Float16)p;
      }
    }

    // ---- P fragments (A-layout) via LDS round-trip ----
    const _Float16* prow = &myP[r * STR];
    v16h pf0 = load_fragA(prow, hi);
    v16h pf1 = load_fragA(prow + 32, hi);

    // ---- row sums of P on the matrix pipe ----
    v8f sums = zero8;
    sums = __builtin_amdgcn_wmma_f32_16x16x32_f16(false, pf0, false, ones, (short)0, sums, false, false);
    sums = __builtin_amdgcn_wmma_f32_16x16x32_f16(false, pf1, false, ones, (short)0, sums, false, false);

#pragma unroll
    for (int j = 0; j < 8; ++j)
      l_run[j] = l_run[j] * sc[j] + sums[j];
#pragma unroll
    for (int g = 0; g < 4; ++g)
#pragma unroll
      for (int j = 0; j < 8; ++j)
        o_acc[g][j] *= sc[j];

    // ---- O += P * V ----
#pragma unroll
    for (int g = 0; g < 4; ++g) {
      const _Float16* vrow = &sVt[(16 * g + r) * STR];
      v16h vf0 = load_fragB(vrow, hi);
      v16h vf1 = load_fragB(vrow + 32, hi);
      o_acc[g] = __builtin_amdgcn_wmma_f32_16x16x32_f16(false, pf0, false, vf0, (short)0, o_acc[g], false, false);
      o_acc[g] = __builtin_amdgcn_wmma_f32_16x16x32_f16(false, pf1, false, vf1, (short)0, o_acc[g], false, false);
    }
  }

  float* outp = Og + ((size_t)b * N_ + qBase + wave * 16) * D_;
#pragma unroll
  for (int j = 0; j < 8; ++j) {
    float inv = 1.0f / l_run[j];
#pragma unroll
    for (int g = 0; g < 4; ++g)
      outp[(size_t)(j + 8 * hi) * D_ + 16 * g + r] = o_acc[g][j] * inv;
  }
}

// ---------------- fallback: fused single-kernel (round-2 version) ------------
__global__ __launch_bounds__(THREADS)
void attn_fused_fallback(const float* __restrict__ Qg, const float* __restrict__ Kg,
                         const float* __restrict__ Vg, float* __restrict__ Og) {
  __shared__ _Float16 sQ[QROWS * STR];
  __shared__ _Float16 sK[KTILE * STR];
  __shared__ _Float16 sVt[D_ * STR];
  __shared__ _Float16 sP[NWAVES * 16 * STR];

  const int tid  = threadIdx.x;
  const int lane = tid & 31;
  const int wave = tid >> 5;
  const int hi   = lane >> 4;
  const int r    = lane & 15;
  const int b     = blockIdx.y;
  const int qBase = blockIdx.x * QROWS;

  const float* Qp = Qg + ((size_t)b * N_ + qBase) * D_;
  const float* Kp = Kg + (size_t)b * M_ * D_;
  const float* Vp = Vg + (size_t)b * M_ * D_;
  const float QSCALE = 0.125f * 1.44269504088896340736f;
  const v8f zero8 = {0.f, 0.f, 0.f, 0.f, 0.f, 0.f, 0.f, 0.f};
  v16h ones;
#pragma unroll
  for (int e = 0; e < 16; ++e) ones[e] = (_Float16)1.0f;

  for (int idx = tid; idx < QROWS * (D_ / 4); idx += THREADS) {
    int row = idx >> 4, c4 = (idx & 15) << 2;
    float4 v = *(const float4*)(Qp + (size_t)row * D_ + c4);
    _Float16* d = &sQ[row * STR + c4];
    d[0] = (_Float16)(v.x * QSCALE); d[1] = (_Float16)(v.y * QSCALE);
    d[2] = (_Float16)(v.z * QSCALE); d[3] = (_Float16)(v.w * QSCALE);
  }
  __syncthreads();
  const _Float16* qrow = &sQ[(wave * 16 + r) * STR];
  v16h qf0 = load_fragA(qrow, hi);
  v16h qf1 = load_fragA(qrow + 32, hi);

  float m_run[8], l_run[8];
  v8f o_acc[4];
#pragma unroll
  for (int j = 0; j < 8; ++j) { m_run[j] = -3.0e38f; l_run[j] = 0.f; }
#pragma unroll
  for (int g = 0; g < 4; ++g) o_acc[g] = zero8;

#pragma unroll 1
  for (int mt = 0; mt < M_; mt += KTILE) {
    __syncthreads();
    for (int idx = tid; idx < KTILE * (D_ / 4); idx += THREADS) {
      int row = idx >> 4, c4 = (idx & 15) << 2;
      float4 v = *(const float4*)(Kp + (size_t)(mt + row) * D_ + c4);
      _Float16* d = &sK[row * STR + c4];
      d[0] = (_Float16)v.x; d[1] = (_Float16)v.y;
      d[2] = (_Float16)v.z; d[3] = (_Float16)v.w;
    }
    for (int idx = tid; idx < KTILE * (D_ / 4); idx += THREADS) {
      int row = idx >> 4, c4 = (idx & 15) << 2;
      float4 v = *(const float4*)(Vp + (size_t)(mt + row) * D_ + c4);
      sVt[(c4 + 0) * STR + row] = (_Float16)v.x;
      sVt[(c4 + 1) * STR + row] = (_Float16)v.y;
      sVt[(c4 + 2) * STR + row] = (_Float16)v.z;
      sVt[(c4 + 3) * STR + row] = (_Float16)v.w;
    }
    __syncthreads();

    v8f s[4];
#pragma unroll
    for (int t = 0; t < 4; ++t) {
      const _Float16* krow = &sK[(16 * t + r) * STR];
      v16h kf0 = load_fragB(krow, hi);
      v16h kf1 = load_fragB(krow + 32, hi);
      v8f c = zero8;
      c = __builtin_amdgcn_wmma_f32_16x16x32_f16(false, qf0, false, kf0, (short)0, c, false, false);
      c = __builtin_amdgcn_wmma_f32_16x16x32_f16(false, qf1, false, kf1, (short)0, c, false, false);
      s[t] = c;
    }
    float mnew[8], sc[8];
#pragma unroll
    for (int j = 0; j < 8; ++j) {
      float v = fmaxf(fmaxf(s[0][j], s[1][j]), fmaxf(s[2][j], s[3][j]));
      v = red_max16(v);
      mnew[j] = fmaxf(m_run[j], v);
      sc[j]   = exp2f(m_run[j] - mnew[j]);
      m_run[j] = mnew[j];
    }
    _Float16* myP = &sP[wave * 16 * STR];
#pragma unroll
    for (int t = 0; t < 4; ++t)
#pragma unroll
      for (int j = 0; j < 8; ++j) {
        float p = exp2f(s[t][j] - mnew[j]);
        myP[(j + 8 * hi) * STR + 16 * t + r] = (_Float16)p;
      }
    const _Float16* prow = &myP[r * STR];
    v16h pf0 = load_fragA(prow, hi);
    v16h pf1 = load_fragA(prow + 32, hi);
    v8f sums = zero8;
    sums = __builtin_amdgcn_wmma_f32_16x16x32_f16(false, pf0, false, ones, (short)0, sums, false, false);
    sums = __builtin_amdgcn_wmma_f32_16x16x32_f16(false, pf1, false, ones, (short)0, sums, false, false);
#pragma unroll
    for (int j = 0; j < 8; ++j) l_run[j] = l_run[j] * sc[j] + sums[j];
#pragma unroll
    for (int g = 0; g < 4; ++g)
#pragma unroll
      for (int j = 0; j < 8; ++j) o_acc[g][j] *= sc[j];
#pragma unroll
    for (int g = 0; g < 4; ++g) {
      const _Float16* vrow = &sVt[(16 * g + r) * STR];
      v16h vf0 = load_fragB(vrow, hi);
      v16h vf1 = load_fragB(vrow + 32, hi);
      o_acc[g] = __builtin_amdgcn_wmma_f32_16x16x32_f16(false, pf0, false, vf0, (short)0, o_acc[g], false, false);
      o_acc[g] = __builtin_amdgcn_wmma_f32_16x16x32_f16(false, pf1, false, vf1, (short)0, o_acc[g], false, false);
    }
  }
  float* outp = Og + ((size_t)b * N_ + qBase + wave * 16) * D_;
#pragma unroll
  for (int j = 0; j < 8; ++j) {
    float inv = 1.0f / l_run[j];
#pragma unroll
    for (int g = 0; g < 4; ++g)
      outp[(size_t)(j + 8 * hi) * D_ + 16 * g + r] = o_acc[g][j] * inv;
  }
}

extern "C" void kernel_launch(void* const* d_in, const int* in_sizes, int n_in,
                              void* d_out, int out_size, void* d_ws, size_t ws_size,
                              hipStream_t stream) {
  (void)in_sizes; (void)n_in; (void)out_size;
  const float* q = (const float*)d_in[0];
  const float* k = (const float*)d_in[1];
  const float* v = (const float*)d_in[2];
  float* out = (float*)d_out;

  const size_t elems = (size_t)B_ * M_ * D_;
  const size_t ws_need = 2 * elems * sizeof(_Float16);  // 8 MB: f16 K + transposed f16 V

  if (d_ws != nullptr && ws_size >= ws_need) {
    _Float16* Kh = (_Float16*)d_ws;
    _Float16* Vt = Kh + elems;
    dim3 pgrid((unsigned)((elems / 4 + 255) / 256), 2, 1);
    attn_prep_kv<<<pgrid, dim3(256), 0, stream>>>(k, v, Kh, Vt);
    attn_main<<<dim3(N_ / QROWS, B_, 1), dim3(THREADS), 0, stream>>>(q, Kh, Vt, out);
  } else {
    attn_fused_fallback<<<dim3(N_ / QROWS, B_, 1), dim3(THREADS), 0, stream>>>(q, k, v, out);
  }
}
